// KPConvLayer_53755810676852
// MI455X (gfx1250) — matrix operand order repaired and checked by reading, not compiled, not run on hardware
//
#include <hip/hip_runtime.h>

// KPConv layer on gfx1250 via v_wmma_f32_16x16x32_f16.
// Inputs (setup_inputs order):
//   0: query_points  [50000,3]  f32
//   1: support_points[50000,3]  f32
//   2: neighbors     [50000,32] i32   (index 50000 == shadow point)
//   3: x             [50000,64] f32
//   4: K_points      [15,3]     f32
//   5: weight        [15,64,64] f32
// Output: [50000,64] f32

typedef __attribute__((ext_vector_type(16))) _Float16 v16h;
typedef __attribute__((ext_vector_type(8)))  _Float16 v8h;
typedef __attribute__((ext_vector_type(8)))  float    v8f;

#define NQ   50000
#define NS   50000
#define MNB  32
#define KPTS 15
#define DIN  64
#define DOUT 64
#define KD   (KPTS * DIN)   // 960 flattened (k,d) inner dimension
#define KDP  (KD + 8)       // wacc row pad: stride 968 halfs = 484 dw -> bank-conflict-free A loads

union V16U { v16h v; v8h h[2]; };

// ---- kernel 0: transpose + f16-convert conv weights: WhT[o][kd] = weight[kd*64+o]
__global__ __launch_bounds__(256)
void kpconv_wcvt(const float* __restrict__ w, _Float16* __restrict__ wht) {
    int i = blockIdx.x * 256 + threadIdx.x;          // 0 .. 61439
    int o  = i / KD;
    int kd = i - o * KD;
    wht[(size_t)o * KD + kd] = (_Float16)w[(size_t)kd * DOUT + o];
}

// ---- main kernel: 1 workgroup = 16 queries, 4 waves of 32
__global__ __launch_bounds__(128)
void kpconv_wmma(const float* __restrict__ qpts,
                 const float* __restrict__ spts,
                 const int*   __restrict__ nbrs,
                 const float* __restrict__ x,
                 const float* __restrict__ kpts,
                 const _Float16* __restrict__ WhT,
                 float* __restrict__ out)
{
    __shared__ _Float16 wgt[16][16][32];    // [q][k(pad16)][m]  A-matrix for step2, 16KB
    __shared__ _Float16 featsT[4][64][32];  // per-wave [d][m]   B-matrix for step2, 16KB
    __shared__ _Float16 wacc[16][KDP];      // [q][k*64+d]+pad   A-matrix for step3, ~30.25KB
    __shared__ float    kps[48];            // kernel point coords

    const int tid  = threadIdx.x;
    const int wv   = tid >> 5;        // wave id 0..3
    const int lane = tid & 31;
    const int lo   = lane & 15;       // matrix row/col within 16
    const int hi   = lane >> 4;       // K-half selector

    // Warm the WGP-local cache with this wave's 30720-byte WhT slice (overlaps
    // with steps 1-2). locality=3 -> WGP scope: pull into all cache levels.
    {
        const char* pbase = (const char*)(WhT + (size_t)(wv * 16) * KD);
        #pragma unroll
        for (int i = 0; i < 8; ++i) {
            const int line = lane + i * 32;          // 240 x 128B lines
            if (line < (16 * KD * 2) / 128)
                __builtin_prefetch(pbase + line * 128, 0, 3);
        }
    }

    if (tid < KPTS * 3) kps[tid] = kpts[tid];
    __syncthreads();

    const int qbase = blockIdx.x * 16;

    for (int iter = 0; iter < 4; ++iter) {
        const int q  = iter * 4 + wv;       // query slot within block
        const int qg = qbase + q;           // global query index

        // ---- step 1: influence weights (lane == neighbor m) ----
        const int  nbr    = nbrs[(size_t)qg * MNB + lane];
        const bool shadow = (nbr >= NS);
        float px, py, pz;
        if (shadow) { px = py = pz = 1.0e6f; }
        else {
            px = spts[nbr * 3 + 0];
            py = spts[nbr * 3 + 1];
            pz = spts[nbr * 3 + 2];
        }
        px -= qpts[qg * 3 + 0];
        py -= qpts[qg * 3 + 1];
        pz -= qpts[qg * 3 + 2];
        #pragma unroll
        for (int k = 0; k < KPTS; ++k) {
            float dx = px - kps[k * 3 + 0];
            float dy = py - kps[k * 3 + 1];
            float dz = pz - kps[k * 3 + 2];
            float wk = fmaxf(1.0f - sqrtf(dx * dx + dy * dy + dz * dz), 0.0f);
            wgt[q][k][lane] = (_Float16)wk;
        }
        wgt[q][15][lane] = (_Float16)0.0f;   // pad K 15 -> 16 with zeros

        // ---- step 1b: gather neighbor features (transposed, f16) ----
        if (shadow) {
            #pragma unroll
            for (int d = 0; d < DIN; ++d) featsT[wv][d][lane] = (_Float16)0.0f;
        } else {
            const float4* row = (const float4*)(x + (size_t)nbr * DIN);
            #pragma unroll
            for (int i = 0; i < DIN / 4; ++i) {
                float4 vv = row[i];
                featsT[wv][4 * i + 0][lane] = (_Float16)vv.x;
                featsT[wv][4 * i + 1][lane] = (_Float16)vv.y;
                featsT[wv][4 * i + 2][lane] = (_Float16)vv.z;
                featsT[wv][4 * i + 3][lane] = (_Float16)vv.w;
            }
        }
        __syncthreads();

        // ---- step 2: weighted[16k x 64d] = wgt[16x32] x feats[32x64] ----
        // A layout (16-bit 16x32): row = lo, K = hi*8 + j (lo 8 halfs), 16 + hi*8 + j (hi 8)
        V16U A;
        A.h[0] = *(const v8h*)&wgt[q][lo][hi * 8];
        A.h[1] = *(const v8h*)&wgt[q][lo][16 + hi * 8];
        #pragma unroll
        for (int t = 0; t < 4; ++t) {            // 4 x 16-wide d tiles
            // B layout (16-bit 32x16): col = lo, K = hi*8 + j (lo 8 halfs), 16 + hi*8 + j (hi 8)
            V16U B;
            const int col = t * 16 + lo;
            B.h[0] = *(const v8h*)&featsT[wv][col][hi * 8];
            B.h[1] = *(const v8h*)&featsT[wv][col][16 + hi * 8];
            v8f c = {};
            c = __builtin_amdgcn_wmma_f32_16x16x32_f16(
                    false, A.v, false, B.v, (short)0, c, false, false);
            // D layout: c[r] = D[r + 8*hi][lo]; row is k -> store k<15 rows to wacc
            #pragma unroll
            for (int r = 0; r < 8; ++r) {
                const int k = r + 8 * hi;
                if (k < KPTS) wacc[q][k * 64 + t * 16 + lo] = (_Float16)c[r];
            }
        }
        __syncthreads();
    }

    // ---- step 3: out[16q x 64o] = wacc[16 x 960] x WhT^T[960 x 64] ----
    // each wave owns one 16-wide o tile; 30 accumulating WMMAs over K=960
    v8f acc = {};
    const _Float16* brow = WhT + (size_t)(wv * 16 + lo) * KD;
    #pragma unroll 5
    for (int ck = 0; ck < KD / 32; ++ck) {
        V16U A, B;
        A.h[0] = *(const v8h*)&wacc[lo][ck * 32 + hi * 8];
        A.h[1] = *(const v8h*)&wacc[lo][ck * 32 + 16 + hi * 8];
        B.h[0] = *(const v8h*)(brow + ck * 32 + hi * 8);
        B.h[1] = *(const v8h*)(brow + ck * 32 + 16 + hi * 8);
        acc = __builtin_amdgcn_wmma_f32_16x16x32_f16(
                  false, A.v, false, B.v, (short)0, acc, false, false);
    }
    // streaming (write-once) output: non-temporal stores keep gather set hot in cache
    #pragma unroll
    for (int r = 0; r < 8; ++r) {
        const int n = qbase + r + 8 * hi;
        __builtin_nontemporal_store(acc[r], &out[(size_t)n * DOUT + wv * 16 + lo]);
    }
}

extern "C" void kernel_launch(void* const* d_in, const int* in_sizes, int n_in,
                              void* d_out, int out_size, void* d_ws, size_t ws_size,
                              hipStream_t stream) {
    const float* qpts = (const float*)d_in[0];
    const float* spts = (const float*)d_in[1];
    const int*   nbrs = (const int*)d_in[2];
    const float* x    = (const float*)d_in[3];
    const float* kpts = (const float*)d_in[4];
    const float* wcon = (const float*)d_in[5];

    _Float16* WhT = (_Float16*)d_ws;       // 64 * 960 f16 = 122880 bytes
    float*    out = (float*)d_out;

    // refresh the f16 transposed weights every call (d_ws is not persistent)
    kpconv_wcvt<<<(DOUT * KD) / 256, 256, 0, stream>>>(wcon, WhT);

    // 50000 queries / 16 per block
    kpconv_wmma<<<NQ / 16, 128, 0, stream>>>(qpts, spts, nbrs, x, kpts, WhT, out);
}
